// CrossModalIntegrationLayer_58772332478711
// MI455X (gfx1250) — compile-verified, matrix operand order checked by reading, not run on hardware
//
#include <hip/hip_runtime.h>
#include <math.h>

// ---------------------------------------------------------------------------
// Problem constants (from reference)
// ---------------------------------------------------------------------------
#define HID 512
#define DI  1024
#define NST 64          // ssm_d_state
#define RK  32          // dt_rank
#define KG  4           // k_group
#define BB  4
#define HH  32
#define WW  32
#define LL  (HH*WW)     // 1024
#define MLPD 2048
#define ROWS (BB*LL)    // 4096 tokens

typedef __attribute__((ext_vector_type(16))) __bf16 v16bf;
typedef __attribute__((ext_vector_type(8)))  float  v8f;

union FragBF { uint4 q[2]; v16bf v; };

__device__ __forceinline__ unsigned short f2bf(float f) {
    unsigned int u = __float_as_uint(f);
    unsigned int r = u + 0x7FFFu + ((u >> 16) & 1u);
    if ((u & 0x7F800000u) == 0x7F800000u) r = u; // inf/nan: truncate
    return (unsigned short)(r >> 16);
}

__device__ __forceinline__ float gelu_exact(float x) {
    return 0.5f * x * (1.0f + erff(x * 0.70710678118654752f));
}

// ---------------------------------------------------------------------------
// Weight packing: f32 (K x N) row-major -> bf16 (N x K) row-major (transpose)
// ---------------------------------------------------------------------------
__global__ void k_pack_transpose(const float* __restrict__ src,
                                 unsigned short* __restrict__ dst,
                                 int K, int N) {
    int idx = blockIdx.x * blockDim.x + threadIdx.x;
    if (idx >= K * N) return;
    int n = idx % N, k = idx / N;
    dst[(size_t)n * K + k] = f2bf(src[idx]);
}

// flat f32 -> bf16 copy (for dt_projs_weight, already (DI x R) = N x K per k)
__global__ void k_pack_copy(const float* __restrict__ src,
                            unsigned short* __restrict__ dst, int n) {
    int idx = blockIdx.x * blockDim.x + threadIdx.x;
    if (idx < n) dst[idx] = f2bf(src[idx]);
}

// gather per-direction dt_rank slices: sp (ROWS x K*(R+N)) -> (K, ROWS, R) bf16
__global__ void k_pack_dtA(const float* __restrict__ sp,
                           unsigned short* __restrict__ dst) {
    int idx = blockIdx.x * blockDim.x + threadIdx.x; // over KG*ROWS*RK
    if (idx >= KG * ROWS * RK) return;
    int k   = idx / (ROWS * RK);
    int rr  = idx % (ROWS * RK);
    int row = rr / RK;
    int r   = rr % RK;
    dst[idx] = f2bf(sp[(size_t)row * (KG * (RK + NST)) + k * RK + r]);
}

// ---------------------------------------------------------------------------
// LayerNorm over HID=512, one wave32 per row, output bf16
// ---------------------------------------------------------------------------
__global__ void k_layernorm_bf16(const float* __restrict__ x,
                                 const float* __restrict__ w,
                                 const float* __restrict__ b,
                                 unsigned short* __restrict__ out, int rows) {
    int wid  = threadIdx.x >> 5;
    int lane = threadIdx.x & 31;
    int row  = blockIdx.x * 8 + wid;
    if (row >= rows) return;
    const float* xr = x + (size_t)row * HID;
    float v[16];
    float s = 0.f;
#pragma unroll
    for (int j = 0; j < 16; ++j) { v[j] = xr[lane + j * 32]; s += v[j]; }
#pragma unroll
    for (int m = 16; m >= 1; m >>= 1) s += __shfl_xor(s, m, 32);
    float mean = s * (1.0f / HID);
    float q = 0.f;
#pragma unroll
    for (int j = 0; j < 16; ++j) { float d = v[j] - mean; q += d * d; }
#pragma unroll
    for (int m = 16; m >= 1; m >>= 1) q += __shfl_xor(q, m, 32);
    float rstd = rsqrtf(q * (1.0f / HID) + 1e-5f);
#pragma unroll
    for (int j = 0; j < 16; ++j) {
        int c = lane + j * 32;
        out[(size_t)row * HID + c] = f2bf((v[j] - mean) * rstd * w[c] + b[c]);
    }
}

// ---------------------------------------------------------------------------
// Generic WMMA BF16 GEMM: out = act(A @ B + bias) + addsrc
//   A  : bf16 (M x Kd) row-major
//   Bt : bf16 (N x Kd) row-major  (i.e. B transposed)
//   block = 256 threads = 8 waves arranged 2(M) x 4(N)
//   wave tile 32x32 = 2x2 WMMA fragments (A/B fragment reuse -> 4 wmma/K-step)
//   block tile 64(M) x 128(N); grid = (N/128, M/64)
// ---------------------------------------------------------------------------
__global__ void k_gemm_bf16(const unsigned short* __restrict__ A,
                            const unsigned short* __restrict__ Bt,
                            int M, int N, int Kd,
                            const float* __restrict__ bias,
                            const float* __restrict__ addsrc,
                            int act,
                            float* __restrict__ outF,
                            unsigned short* __restrict__ outBF) {
    const int lane   = threadIdx.x & 31;
    const int waveId = threadIdx.x >> 5;
    const int tileM  = blockIdx.y * 64  + (waveId >> 2) * 32;
    const int tileN  = blockIdx.x * 128 + (waveId & 3) * 32;

    const int mfrag = lane & 15;             // fragment row (A) / row (Bt)
    const int khalf = (lane >> 4) * 8;       // K-half split across lane groups

    const unsigned short* arow0 = A  + (size_t)(tileM + mfrag) * Kd;
    const unsigned short* arow1 = A  + (size_t)(tileM + 16 + mfrag) * Kd;
    const unsigned short* brow0 = Bt + (size_t)(tileN + mfrag) * Kd;
    const unsigned short* brow1 = Bt + (size_t)(tileN + 16 + mfrag) * Kd;

    v8f acc00 = {}, acc01 = {}, acc10 = {}, acc11 = {};
    for (int k0 = 0; k0 < Kd; k0 += 32) {
        FragBF fa0, fa1, fb0, fb1;
        fa0.q[0] = *reinterpret_cast<const uint4*>(arow0 + k0 + khalf);
        fa0.q[1] = *reinterpret_cast<const uint4*>(arow0 + k0 + 16 + khalf);
        fa1.q[0] = *reinterpret_cast<const uint4*>(arow1 + k0 + khalf);
        fa1.q[1] = *reinterpret_cast<const uint4*>(arow1 + k0 + 16 + khalf);
        fb0.q[0] = *reinterpret_cast<const uint4*>(brow0 + k0 + khalf);
        fb0.q[1] = *reinterpret_cast<const uint4*>(brow0 + k0 + 16 + khalf);
        fb1.q[0] = *reinterpret_cast<const uint4*>(brow1 + k0 + khalf);
        fb1.q[1] = *reinterpret_cast<const uint4*>(brow1 + k0 + 16 + khalf);
        if (k0 + 32 < Kd) {
            __builtin_prefetch(arow0 + k0 + 32, 0, 3);   // global_prefetch_b8
            __builtin_prefetch(arow1 + k0 + 32, 0, 3);
            __builtin_prefetch(brow0 + k0 + 32, 0, 3);
            __builtin_prefetch(brow1 + k0 + 32, 0, 3);
        }
        acc00 = __builtin_amdgcn_wmma_f32_16x16x32_bf16(
            false, fa0.v, false, fb0.v, (short)0, acc00, false, false);
        acc01 = __builtin_amdgcn_wmma_f32_16x16x32_bf16(
            false, fa0.v, false, fb1.v, (short)0, acc01, false, false);
        acc10 = __builtin_amdgcn_wmma_f32_16x16x32_bf16(
            false, fa1.v, false, fb0.v, (short)0, acc10, false, false);
        acc11 = __builtin_amdgcn_wmma_f32_16x16x32_bf16(
            false, fa1.v, false, fb1.v, (short)0, acc11, false, false);
    }

    // epilogue: lane holds rows (8*(lane>>4)+i) and col (lane&15) of each subtile
    const int rb = (lane >> 4) * 8;
    const int cl = lane & 15;
#pragma unroll
    for (int sub = 0; sub < 4; ++sub) {
        const v8f* accp = (sub == 0) ? &acc00 : (sub == 1) ? &acc01
                        : (sub == 2) ? &acc10 : &acc11;
        const int rowB = tileM + ((sub >> 1) ? 16 : 0) + rb;
        const int col  = tileN + ((sub & 1) ? 16 : 0) + cl;
        float bv = bias ? bias[col] : 0.0f;
#pragma unroll
        for (int i = 0; i < 8; ++i) {
            int row = rowB + i;
            float v = (*accp)[i] + bv;
            if (act == 1) v = gelu_exact(v);
            if (addsrc) v += addsrc[(size_t)row * N + col];
            if (outF)  outF [(size_t)row * N + col] = v;
            if (outBF) outBF[(size_t)row * N + col] = f2bf(v);
        }
    }
}

// ---------------------------------------------------------------------------
// Depthwise 3x3 conv (SAME) + bias + SiLU.
// x lives in xz (ROWS x 2*DI) columns [0,DI). Output xa f32 + bf16, (ROWS x DI)
// ---------------------------------------------------------------------------
__global__ void k_conv_silu(const float* __restrict__ xz,
                            const float* __restrict__ cw,
                            const float* __restrict__ cb,
                            float* __restrict__ xa,
                            unsigned short* __restrict__ xabf) {
    int idx = blockIdx.x * blockDim.x + threadIdx.x; // over BB*LL*DI
    if (idx >= BB * LL * DI) return;
    int c = idx % DI;
    int l = (idx / DI) % LL;
    int b = idx / (DI * LL);
    int h = l >> 5, w = l & 31;
    float acc = cb[c];
#pragma unroll
    for (int dh = -1; dh <= 1; ++dh) {
#pragma unroll
        for (int dw = -1; dw <= 1; ++dw) {
            int hh = h + dh, ww = w + dw;
            if (hh >= 0 && hh < HH && ww >= 0 && ww < WW) {
                float xv = xz[((size_t)(b * LL + hh * WW + ww)) * (2 * DI) + c];
                acc += xv * cw[c * 9 + (dh + 1) * 3 + (dw + 1)];
            }
        }
    }
    float s = acc / (1.0f + __expf(-acc)); // SiLU
    xa[(size_t)(b * LL + l) * DI + c] = s;
    xabf[(size_t)(b * LL + l) * DI + c] = f2bf(s);
}

// ---------------------------------------------------------------------------
// Selective scan. One thread per (b,k,d) channel; h[NST] in registers.
// grid = BB*KG*(DI/256) blocks of 256 threads.
//   u_t from xa with per-direction index remap; delta/B/C indexed by raster t.
//   ys[k,b,t,d] = <h_t, C_t> + Ds[k,d]*u_t
// ---------------------------------------------------------------------------
__global__ void k_scan(const float* __restrict__ xa,      // (ROWS x DI) spatial
                       const float* __restrict__ dts,     // (KG, ROWS, DI)
                       const float* __restrict__ sp,      // (ROWS x KG*(R+N))
                       const float* __restrict__ cs,      // (ROWS x KG*N)
                       const float* __restrict__ dt_bias, // (KG, DI)
                       const float* __restrict__ A_logs,  // (KG*DI, NST)
                       const float* __restrict__ Ds,      // (KG*DI)
                       float* __restrict__ ys) {          // (KG, BB, LL, DI)
    const int tid = threadIdx.x;
    const int blk = blockIdx.x;               // over BB*KG*4
    const int b   = blk / (KG * 4);
    const int rem = blk % (KG * 4);
    const int k   = rem / 4;
    const int d   = (rem % 4) * 256 + tid;

    __shared__ float sB[NST];
    __shared__ float sC[NST];

    float h[NST];
    float Ar[NST];
#pragma unroll
    for (int n = 0; n < NST; ++n) {
        h[n]  = 0.0f;
        Ar[n] = -__expf(A_logs[(size_t)(k * DI + d) * NST + n]);
    }
    const float bias = dt_bias[k * DI + d];
    const float dsv  = Ds[k * DI + d];
    const int spStride = KG * (RK + NST);   // 384
    const int csStride = KG * NST;          // 256

    for (int t = 0; t < LL; ++t) {
        // stage B_t, C_t (shared over d) into LDS
        if (tid < NST)
            sB[tid] = sp[(size_t)(b * LL + t) * spStride + KG * RK + k * NST + tid];
        else if (tid < 2 * NST)
            sC[tid - NST] = cs[(size_t)(b * LL + t) * csStride + k * NST + (tid - NST)];
        __syncthreads();

        // direction-ordered input index
        int lp;
        if (k == 0)      lp = t;
        else if (k == 1) lp = (t & 31) * 32 + (t >> 5);
        else if (k == 2) lp = LL - 1 - t;
        else { int t2 = LL - 1 - t; lp = (t2 & 31) * 32 + (t2 >> 5); }

        float u = xa[(size_t)(b * LL + lp) * DI + d];
        float dv = dts[(size_t)((k * ROWS) + (b * LL + t)) * DI + d] + bias;
        float delta = (dv > 20.0f) ? dv : log1pf(__expf(dv));
        float du = delta * u;

        float y = 0.0f;
#pragma unroll
        for (int n = 0; n < NST; ++n) {
            float g = __expf(delta * Ar[n]);
            h[n] = g * h[n] + du * sB[n];
            y += h[n] * sC[n];
        }
        ys[(size_t)(((k * BB + b) * LL + t)) * DI + d] = y + dsv * u;
        __syncthreads();
    }
}

// ---------------------------------------------------------------------------
// Combine: y_pre = (sum_k ys) * z  -> bf16  (ROWS x DI)
// z = xz[:, DI:2*DI]
// ---------------------------------------------------------------------------
__global__ void k_combine(const float* __restrict__ ys,
                          const float* __restrict__ xz,
                          unsigned short* __restrict__ ypre) {
    int idx = blockIdx.x * blockDim.x + threadIdx.x; // over ROWS*DI
    if (idx >= ROWS * DI) return;
    int d = idx & (DI - 1);
    int r = idx >> 10;          // token row = b*LL + l
    int b = r >> 10;
    int l = r & (LL - 1);
    float s = 0.0f;
#pragma unroll
    for (int k = 0; k < KG; ++k)
        s += ys[(size_t)(((k * BB + b) * LL + l)) * DI + d];
    float z = xz[(size_t)r * (2 * DI) + DI + d];
    ypre[idx] = f2bf(s * z);
}

// ---------------------------------------------------------------------------
// Host-side launcher
// ---------------------------------------------------------------------------
static inline size_t alignup(size_t x) { return (x + 255) & ~(size_t)255; }

extern "C" void kernel_launch(void* const* d_in, const int* in_sizes, int n_in,
                              void* d_out, int out_size, void* d_ws, size_t ws_size,
                              hipStream_t stream) {
    (void)in_sizes; (void)n_in; (void)out_size; (void)ws_size;
    const float* content   = (const float*)d_in[0];
    const float* style     = (const float*)d_in[1];
    const float* norm1_w   = (const float*)d_in[2];
    const float* norm1_b   = (const float*)d_in[3];
    const float* in_proj_w = (const float*)d_in[4];
    const float* conv_w    = (const float*)d_in[5];
    const float* conv_b    = (const float*)d_in[6];
    const float* style_w   = (const float*)d_in[7];
    const float* cont_w    = (const float*)d_in[8];
    const float* dt_w      = (const float*)d_in[9];
    const float* dt_b      = (const float*)d_in[10];
    const float* A_logs    = (const float*)d_in[11];
    const float* Ds        = (const float*)d_in[12];
    const float* out_w     = (const float*)d_in[13];
    const float* norm2_w   = (const float*)d_in[14];
    const float* norm2_b   = (const float*)d_in[15];
    const float* mlp_w1    = (const float*)d_in[16];
    const float* mlp_b1    = (const float*)d_in[17];
    const float* mlp_w2    = (const float*)d_in[18];
    const float* mlp_b2    = (const float*)d_in[19];
    float* outp = (float*)d_out;

    // ---- scratch arena ----
    char* p = (char*)d_ws;
    auto alloc = [&](size_t bytes) -> void* { void* r = p; p += alignup(bytes); return r; };

    unsigned short* cn_bf   = (unsigned short*)alloc((size_t)ROWS * HID * 2);
    unsigned short* sn_bf   = (unsigned short*)alloc((size_t)ROWS * HID * 2);
    unsigned short* wInT    = (unsigned short*)alloc((size_t)(2 * DI) * HID * 2);
    unsigned short* wStyT   = (unsigned short*)alloc((size_t)(KG * (RK + NST)) * HID * 2);
    unsigned short* wConT   = (unsigned short*)alloc((size_t)(KG * NST) * DI * 2);
    unsigned short* wOutT   = (unsigned short*)alloc((size_t)HID * DI * 2);
    unsigned short* wM1T    = (unsigned short*)alloc((size_t)MLPD * HID * 2);
    unsigned short* wM2T    = (unsigned short*)alloc((size_t)HID * MLPD * 2);
    unsigned short* wDt     = (unsigned short*)alloc((size_t)KG * DI * RK * 2);
    float*          xz      = (float*)alloc((size_t)ROWS * 2 * DI * 4);
    float*          xa      = (float*)alloc((size_t)ROWS * DI * 4);
    unsigned short* xa_bf   = (unsigned short*)alloc((size_t)ROWS * DI * 2);
    float*          sp      = (float*)alloc((size_t)ROWS * KG * (RK + NST) * 4);
    unsigned short* dtA     = (unsigned short*)alloc((size_t)KG * ROWS * RK * 2);
    float*          dts     = (float*)alloc((size_t)KG * ROWS * DI * 4);
    float*          cs      = (float*)alloc((size_t)ROWS * KG * NST * 4);
    float*          ysb     = (float*)alloc((size_t)KG * BB * LL * DI * 4);
    unsigned short* ypre    = (unsigned short*)alloc((size_t)ROWS * DI * 2);
    float*          x1      = (float*)alloc((size_t)ROWS * HID * 4);
    unsigned short* xn      = (unsigned short*)alloc((size_t)ROWS * HID * 2);
    unsigned short* m1      = (unsigned short*)alloc((size_t)ROWS * MLPD * 2);

    const int T = 256;
    auto g1 = [&](int n) { return dim3((n + T - 1) / T); };
    auto gemmGrid = [&](int N, int M) { return dim3(N / 128, M / 64); };

    // ---- 1. weight packs (transpose K x N -> N x K, bf16) ----
    k_pack_transpose<<<g1(HID * 2 * DI), T, 0, stream>>>(in_proj_w, wInT, HID, 2 * DI);
    k_pack_transpose<<<g1(HID * KG * (RK + NST)), T, 0, stream>>>(style_w, wStyT, HID, KG * (RK + NST));
    k_pack_transpose<<<g1(DI * KG * NST), T, 0, stream>>>(cont_w, wConT, DI, KG * NST);
    k_pack_transpose<<<g1(DI * HID), T, 0, stream>>>(out_w, wOutT, DI, HID);
    k_pack_transpose<<<g1(HID * MLPD), T, 0, stream>>>(mlp_w1, wM1T, HID, MLPD);
    k_pack_transpose<<<g1(MLPD * HID), T, 0, stream>>>(mlp_w2, wM2T, MLPD, HID);
    k_pack_copy<<<g1(KG * DI * RK), T, 0, stream>>>(dt_w, wDt, KG * DI * RK);

    // ---- 2. LayerNorm content & style -> bf16 (norm1 shared) ----
    k_layernorm_bf16<<<dim3(ROWS / 8), T, 0, stream>>>(content, norm1_w, norm1_b, cn_bf, ROWS);
    k_layernorm_bf16<<<dim3(ROWS / 8), T, 0, stream>>>(style, norm1_w, norm1_b, sn_bf, ROWS);

    // ---- 3. in_proj: (ROWS x HID) @ (HID x 2DI) -> xz f32 ----
    k_gemm_bf16<<<gemmGrid(2 * DI, ROWS), T, 0, stream>>>(
        cn_bf, wInT, ROWS, 2 * DI, HID, nullptr, nullptr, 0, xz, nullptr);

    // ---- 4. depthwise conv + SiLU -> xa (f32 + bf16) ----
    k_conv_silu<<<g1(BB * LL * DI), T, 0, stream>>>(xz, conv_w, conv_b, xa, xa_bf);

    // ---- 5. style_proj -> sp f32 (ROWS x 384) ----
    k_gemm_bf16<<<gemmGrid(KG * (RK + NST), ROWS), T, 0, stream>>>(
        sn_bf, wStyT, ROWS, KG * (RK + NST), HID, nullptr, nullptr, 0, sp, nullptr);

    // ---- 6. content_proj: xa @ (DI x 256) -> cs f32 ----
    k_gemm_bf16<<<gemmGrid(KG * NST, ROWS), T, 0, stream>>>(
        xa_bf, wConT, ROWS, KG * NST, DI, nullptr, nullptr, 0, cs, nullptr);

    // ---- 7. dt projections: per k, (ROWS x R) @ (R x DI) -> dts f32 ----
    k_pack_dtA<<<g1(KG * ROWS * RK), T, 0, stream>>>(sp, dtA);
    for (int k = 0; k < KG; ++k) {
        k_gemm_bf16<<<gemmGrid(DI, ROWS), T, 0, stream>>>(
            dtA + (size_t)k * ROWS * RK, wDt + (size_t)k * DI * RK,
            ROWS, DI, RK, nullptr, nullptr, 0,
            dts + (size_t)k * ROWS * DI, nullptr);
    }

    // ---- 8. selective scan -> ysb ----
    k_scan<<<dim3(BB * KG * (DI / 256)), T, 0, stream>>>(
        xa, dts, sp, cs, dt_b, A_logs, Ds, ysb);

    // ---- 9. combine directions, gate by z -> ypre bf16 ----
    k_combine<<<g1(ROWS * DI), T, 0, stream>>>(ysb, xz, ypre);

    // ---- 10. out_proj + residual(content) -> x1 f32 ----
    k_gemm_bf16<<<gemmGrid(HID, ROWS), T, 0, stream>>>(
        ypre, wOutT, ROWS, HID, DI, nullptr, content, 0, x1, nullptr);

    // ---- 11. LN2 -> xn bf16 ----
    k_layernorm_bf16<<<dim3(ROWS / 8), T, 0, stream>>>(x1, norm2_w, norm2_b, xn, ROWS);

    // ---- 12. MLP1: GELU(xn @ w1 + b1) -> m1 bf16 ----
    k_gemm_bf16<<<gemmGrid(MLPD, ROWS), T, 0, stream>>>(
        xn, wM1T, ROWS, MLPD, HID, mlp_b1, nullptr, 1, nullptr, m1);

    // ---- 13. MLP2: m1 @ w2 + b2 + x1 -> d_out f32 ----
    k_gemm_bf16<<<gemmGrid(HID, ROWS), T, 0, stream>>>(
        m1, wM2T, ROWS, HID, MLPD, mlp_b2, x1, 0, outp, nullptr);
}
